// HeteroGNN_59854664237128
// MI455X (gfx1250) — matrix operand order repaired and checked by reading, not compiled, not run on hardware
//
#include <hip/hip_runtime.h>

#define NP 100000
#define NA 50000
#define FP 128
#define FA 64
#define HID 256
#define OUTD 64
#define NE 300000
#define MT 2   // 16-row M tiles per wave
#define NT 4   // 16-wide N tiles per wave

typedef __attribute__((ext_vector_type(16))) __bf16 v16bf;
typedef __attribute__((ext_vector_type(8)))  float  v8f;

// ---------------------------------------------------------------------------
// Pack W[K,N] (f32, row-major) into WMMA B-fragment-ready bf16:
// fragment f = nTile*(K/32)+kTile; lane l owns col nTile*16+(l&15),
// K = kTile*32 + 16*(l>>4) + j, j=0..15, stored contiguously (16 bf16 = 32B).
// ---------------------------------------------------------------------------
__global__ void pack_w_kernel(const float* __restrict__ W, __bf16* __restrict__ Wp,
                              int K, int N)
{
  const int kTiles = K / 32;
  const int nTiles = N / 16;
  const int t    = blockIdx.x * blockDim.x + threadIdx.x;
  const int lane = t & 31;
  const int frag = t >> 5;
  if (frag >= kTiles * nTiles) return;
  const int ntile = frag / kTiles;
  const int ktile = frag % kTiles;
  const int col = ntile * 16 + (lane & 15);
  const int kb  = ktile * 32 + (lane >> 4) * 16;
  __bf16* out = Wp + ((size_t)frag * 32 + lane) * 16;
  #pragma unroll
  for (int j = 0; j < 16; ++j)
    out[j] = (__bf16)W[(size_t)(kb + j) * N + col];
}

// ---------------------------------------------------------------------------
// Cout[M,N] = (Cin ? Cin : 0) + (bias ? bias : 0) + A[M,K] @ W[K,N], opt. relu
// bf16 WMMA 16x16x32; wave owns a 32x64 output span (2x4 accumulators): each
// B fragment (one 32B vector load, 32-bit offsets) feeds 2 WMMAs.
// ---------------------------------------------------------------------------
__global__ void __launch_bounds__(128)
wmma_gemm_kernel(const float* __restrict__ A, const __bf16* __restrict__ Wp,
                 const float* __restrict__ bias, const float* __restrict__ Cin,
                 float* __restrict__ Cout, int M, int K, int N, int relu)
{
  const int lane   = threadIdx.x;                // 0..31 (wave32)
  const int mTiles = M / 16;
  const int pairs  = (mTiles + MT - 1) / MT;
  const int p      = blockIdx.y * 4 + threadIdx.y;
  if (p >= pairs) return;                        // wave-uniform exit (EXEC all-1 for WMMA)
  int mT[MT];
  mT[0] = p * MT;
  mT[1] = (p * MT + 1 < mTiles) ? (p * MT + 1) : (mTiles - 1);  // clamp: benign dup in-wave
  const int colBase = blockIdx.x * (16 * NT);
  const int nlo = lane & 15;
  const int hi  = lane >> 4;                     // 0 or 1
  const int kTiles = K / 32;

  v8f acc[MT][NT];
  #pragma unroll
  for (int m = 0; m < MT; ++m) {                 // C/D layout: M = v + 8*hi, N = nlo
    const int row0 = mT[m] * 16;
    #pragma unroll
    for (int t = 0; t < NT; ++t) {
      const int col = colBase + t * 16 + nlo;
      #pragma unroll
      for (int v = 0; v < 8; ++v) {
        const int row = row0 + v + hi * 8;
        float c0 = Cin ? Cin[(size_t)row * N + col] : 0.0f;
        if (bias) c0 += bias[col];
        acc[m][t][v] = c0;
      }
    }
  }

  // 32-bit element offsets into packed weights -> global (saddr+voffset) loads
  unsigned wOff[NT];
  #pragma unroll
  for (int t = 0; t < NT; ++t)
    wOff[t] = ((unsigned)(colBase / 16 + t) * (unsigned)kTiles * 32u + (unsigned)lane) * 16u;

  const float* Arow0 = A + (size_t)(mT[0] * 16 + nlo) * K;
  const float* Arow1 = A + (size_t)(mT[1] * 16 + nlo) * K;

  for (int k0 = 0; k0 < K; k0 += 32) {
    // ---- A fragments: lane row = nlo; K = {ka..ka+15}, ka = k0+16*hi
    const int ka = k0 + hi * 16;
    v16bf af[MT];
    #pragma unroll
    for (int m = 0; m < MT; ++m) {
      const float* Ar = (m == 0) ? Arow0 : Arow1;
      const float4 a0 = *(const float4*)(Ar + ka);
      const float4 a1 = *(const float4*)(Ar + ka + 4);
      const float4 a2 = *(const float4*)(Ar + ka + 8);
      const float4 a3 = *(const float4*)(Ar + ka + 12);
      af[m][0]  = (__bf16)a0.x; af[m][1]  = (__bf16)a0.y;
      af[m][2]  = (__bf16)a0.z; af[m][3]  = (__bf16)a0.w;
      af[m][4]  = (__bf16)a1.x; af[m][5]  = (__bf16)a1.y;
      af[m][6]  = (__bf16)a1.z; af[m][7]  = (__bf16)a1.w;
      af[m][8]  = (__bf16)a2.x; af[m][9]  = (__bf16)a2.y;
      af[m][10] = (__bf16)a2.z; af[m][11] = (__bf16)a2.w;
      af[m][12] = (__bf16)a3.x; af[m][13] = (__bf16)a3.y;
      af[m][14] = (__bf16)a3.z; af[m][15] = (__bf16)a3.w;
    }

    #pragma unroll
    for (int t = 0; t < NT; ++t) {
      const v16bf bf = *(const v16bf*)(Wp + wOff[t]);   // 32B contiguous -> 2x b128
      wOff[t] += 512u;
      #pragma unroll
      for (int m = 0; m < MT; ++m)
        acc[m][t] = __builtin_amdgcn_wmma_f32_16x16x32_bf16(false, af[m], false, bf,
                                                            (short)0, acc[m][t],
                                                            false, false);
    }
  }

  #pragma unroll
  for (int m = 0; m < MT; ++m) {
    const int row0 = mT[m] * 16;
    #pragma unroll
    for (int t = 0; t < NT; ++t) {
      const int col = colBase + t * 16 + nlo;
      #pragma unroll
      for (int v = 0; v < 8; ++v) {
        const int row = row0 + v + hi * 8;
        float o = acc[m][t][v];
        if (relu) o = fmaxf(o, 0.0f);
        Cout[(size_t)row * N + col] = o;
      }
    }
  }
}

// ---------------------------------------------------------------------------
// Aggregation helpers
// ---------------------------------------------------------------------------
__global__ void fill_kernel(float* __restrict__ p, long n) {
  long t = (long)blockIdx.x * blockDim.x + threadIdx.x;
  if (t < n) p[t] = 0.0f;
}

// edge: src ids at [0,E), dst ids at [E,2E) (int64). One thread = one float4 chunk.
__global__ void scatter_add_kernel(const float* __restrict__ feat,
                                   const long long* __restrict__ edge,
                                   float* __restrict__ agg, int E, int nfeat)
{
  const int chunks = nfeat / 4;
  long t = (long)blockIdx.x * blockDim.x + threadIdx.x;
  if (t >= (long)E * chunks) return;
  const int e = (int)(t / chunks);
  const int c = (int)(t % chunks) * 4;
  const int s = (int)edge[e];
  const int d = (int)edge[E + e];
  const float4 v = *(const float4*)(feat + (size_t)s * nfeat + c);
  float* o = agg + (size_t)d * nfeat + c;
  atomicAdd(o + 0, v.x);
  atomicAdd(o + 1, v.y);
  atomicAdd(o + 2, v.z);
  atomicAdd(o + 3, v.w);
}

__global__ void count_kernel(const long long* __restrict__ edge,
                             float* __restrict__ cnt, int E)
{
  long t = (long)blockIdx.x * blockDim.x + threadIdx.x;
  if (t < E) atomicAdd(&cnt[(int)edge[E + t]], 1.0f);
}

__global__ void divide_kernel(float* __restrict__ agg,
                              const float* __restrict__ cnt, long n, int nfeat)
{
  long t = (long)blockIdx.x * blockDim.x + threadIdx.x;
  if (t < n) agg[t] *= __frcp_rn(fmaxf(cnt[t / nfeat], 1.0f));
}

// ---------------------------------------------------------------------------
extern "C" void kernel_launch(void* const* d_in, const int* in_sizes, int n_in,
                              void* d_out, int out_size, void* d_ws, size_t ws_size,
                              hipStream_t stream) {
  (void)in_sizes; (void)n_in; (void)out_size; (void)ws_size;
  const float*     x_p  = (const float*)d_in[0];
  const float*     x_a  = (const float*)d_in[1];
  const long long* e_wr = (const long long*)d_in[2];   // writes:  author -> paper
  const long long* e_wn = (const long long*)d_in[3];   // written: paper  -> author
  const long long* e_ct = (const long long*)d_in[4];   // cites:   paper  -> paper
  const float* blp  = (const float*)d_in[6];
  const float* bla  = (const float*)d_in[8];
  const float* bout = (const float*)d_in[28];

  // workspace layout (floats), end of float region is 32B-aligned
  float* yp   = (float*)d_ws;
  float* ya   = yp  + (long)NP * HID;
  float* npb  = ya  + (long)NA * HID;
  float* nab  = npb + (long)NP * HID;
  float* agg  = nab + (long)NA * HID;      // reused (paper-sized covers author too)
  float* cnt  = agg + (long)NP * HID;      // NP floats, reused
  __bf16* wp_base = (__bf16*)(cnt + NP);   // packed weights (bf16), ~1.7MB
  size_t wp_off = 0;

  auto packw = [&](int idx, int K, int N) -> const __bf16* {
    __bf16* p = wp_base + wp_off;
    wp_off += (size_t)K * N;
    const int frags = (K / 32) * (N / 16);
    pack_w_kernel<<<(unsigned)((frags * 32 + 255) / 256), 256, 0, stream>>>(
        (const float*)d_in[idx], p, K, N);
    return p;
  };
  auto gemm = [&](const float* A, const __bf16* Wp, const float* bias,
                  const float* Cin, float* Cout, int M, int K, int N, int relu) {
    const int pairs = (M / 16 + MT - 1) / MT;
    dim3 grid((unsigned)(N / (16 * NT)), (unsigned)((pairs + 3) / 4));
    dim3 block(32, 4);
    wmma_gemm_kernel<<<grid, block, 0, stream>>>(A, Wp, bias, Cin, Cout, M, K, N, relu);
  };
  auto aggregate = [&](const float* feat, const long long* edge, int ndst) {
    const long nagg = (long)ndst * HID;
    fill_kernel<<<(unsigned)((nagg + 255) / 256), 256, 0, stream>>>(agg, nagg);
    fill_kernel<<<(unsigned)((ndst + 255) / 256), 256, 0, stream>>>(cnt, ndst);
    scatter_add_kernel<<<(unsigned)(((long)NE * (HID / 4) + 255) / 256), 256, 0, stream>>>(
        feat, edge, agg, NE, HID);
    count_kernel<<<(NE + 255) / 256, 256, 0, stream>>>(edge, cnt, NE);
    divide_kernel<<<(unsigned)((nagg + 255) / 256), 256, 0, stream>>>(agg, cnt, nagg, HID);
  };

  // ---- pack all GEMM weights once (bf16 fragment layout) ----
  const __bf16* Wlp_p  = packw(5,  FP,  HID);
  const __bf16* Wla_p  = packw(7,  FA,  HID);
  const __bf16* Lw[2][6];   // per layer: writes{Wl,Wr}, written{Wl,Wr}, cites{Wl,Wr}
  for (int l = 0; l < 2; ++l) {
    const int b = 9 + l * 9;
    Lw[l][0] = packw(b + 0, HID, HID);   // writes.Wl
    Lw[l][1] = packw(b + 2, HID, HID);   // writes.Wr
    Lw[l][2] = packw(b + 3, HID, HID);   // written.Wl
    Lw[l][3] = packw(b + 5, HID, HID);   // written.Wr
    Lw[l][4] = packw(b + 6, HID, HID);   // cites.Wl
    Lw[l][5] = packw(b + 8, HID, HID);   // cites.Wr
  }
  const __bf16* Wout_p = packw(27, HID, OUTD);

  // ---- input projections + relu ----
  gemm(x_p, Wlp_p, blp, nullptr, yp, NP, FP, HID, 1);
  gemm(x_a, Wla_p, bla, nullptr, ya, NA, FA, HID, 1);

  float* cur_p = yp;  float* nxt_p = npb;
  float* cur_a = ya;  float* nxt_a = nab;
  for (int l = 0; l < 2; ++l) {
    const int b = 9 + l * 9;
    const float* blW = (const float*)d_in[b + 1];
    const float* blN = (const float*)d_in[b + 4];
    const float* blC = (const float*)d_in[b + 7];

    // paper update: writes (author->paper) + cites (paper->paper)
    aggregate(cur_a, e_wr, NP);
    gemm(agg,   Lw[l][0], blW,     nullptr, nxt_p, NP, HID, HID, 0);
    gemm(cur_p, Lw[l][1], nullptr, nxt_p,   nxt_p, NP, HID, HID, 0);
    aggregate(cur_p, e_ct, NP);
    gemm(agg,   Lw[l][4], blC,     nxt_p,   nxt_p, NP, HID, HID, 0);
    gemm(cur_p, Lw[l][5], nullptr, nxt_p,   nxt_p, NP, HID, HID, 0);

    // author update: written (paper->author)
    aggregate(cur_p, e_wn, NA);
    gemm(agg,   Lw[l][2], blN,     nullptr, nxt_a, NA, HID, HID, 0);
    gemm(cur_a, Lw[l][3], nullptr, nxt_a,   nxt_a, NA, HID, HID, 0);

    float* t;
    t = cur_p; cur_p = nxt_p; nxt_p = t;
    t = cur_a; cur_a = nxt_a; nxt_a = t;
  }

  // final projection on paper nodes -> d_out [NP, 64]
  gemm(cur_p, Wout_p, bout, nullptr, (float*)d_out, NP, HID, OUTD, 0);
}
